// LearningModel_2276332667028
// MI455X (gfx1250) — matrix operand order repaired and checked by reading, not compiled, not run on hardware
//
#include <hip/hip_runtime.h>
#include <math.h>

#define N_CLAUSES 200000
#define N_FEAT    64
#define HIDDEN    128
#define N_STEPS   512
#define SPLIT     4            // segments per step for the masked reduction

typedef __attribute__((ext_vector_type(16))) __bf16          v16bf;
typedef __attribute__((ext_vector_type(16))) unsigned short  v16us;
typedef __attribute__((ext_vector_type(8)))  float           v8f;

__device__ __forceinline__ unsigned short f2bf(float f) {
  unsigned u = __float_as_uint(f);
  unsigned r = u + 0x7FFFu + ((u >> 16) & 1u);   // round-to-nearest-even
  return (unsigned short)(r >> 16);
}
__device__ __forceinline__ unsigned f2key(float f) {   // order-preserving float->uint
  unsigned u = __float_as_uint(f);
  return (u & 0x80000000u) ? ~u : (u | 0x80000000u);
}
__device__ __forceinline__ float key2f(unsigned k) {
  unsigned u = (k & 0x80000000u) ? (k ^ 0x80000000u) : ~k;
  return __uint_as_float(u);
}

// ---------------------------------------------------------------- prep
// v = W2 @ key_w ; c = b2 . key_w ; W1t_bf16[n][k] = bf16(W1[k][n])
__global__ void prep_kernel(const float* __restrict__ W1, const float* __restrict__ b1,
                            const float* __restrict__ W2, const float* __restrict__ b2,
                            const float* __restrict__ key_w,
                            unsigned* hdr, unsigned short* __restrict__ W1t,
                            float2* __restrict__ bv) {
  int t = threadIdx.x;                       // 0..127
  float v = 0.f;
  for (int j = 0; j < HIDDEN; ++j) v += W2[t * HIDDEN + j] * key_w[j];
  bv[t] = make_float2(b1[t], v);
  for (int k = 0; k < N_FEAT; ++k)
    W1t[t * N_FEAT + k] = f2bf(W1[k * HIDDEN + t]);
  if (t == 0) {
    float c = 0.f;
    for (int j = 0; j < HIDDEN; ++j) c += b2[j] * key_w[j];
    hdr[0] = 0u;                 // running-max key (ordered uint)
    ((float*)hdr)[1] = c;        // bias constant
    hdr[2] = 0u;                 // proof count (set by compact_kernel)
  }
}

// ---------------------------------------------------------------- logits (WMMA)
// One wave -> 16 clause rows. 8 N-tiles x (K=64 as 2x K32) = 16 wmma/wave.
__global__ __launch_bounds__(256)
void logits_kernel(const float* __restrict__ features,
                   const unsigned short* __restrict__ W1t_g,
                   const float2* __restrict__ bv_g,
                   const unsigned* hdr,
                   float* __restrict__ logits,
                   unsigned* maxbits) {
  __shared__ unsigned short sW1t[HIDDEN * N_FEAT];   // 16 KB, bf16 bits, [n][k]
  __shared__ float2 sBV[HIDDEN];
  int tid = threadIdx.x;
  {
    const unsigned* src = (const unsigned*)W1t_g;
    unsigned* dst = (unsigned*)sW1t;
    for (int i = tid; i < HIDDEN * N_FEAT / 2; i += 256) dst[i] = src[i];
    for (int i = tid; i < HIDDEN; i += 256) sBV[i] = bv_g[i];
  }
  __syncthreads();

  int wave = tid >> 5;
  int lane = tid & 31;
  int hv   = lane >> 4;        // half-wave (ISA A/B striping)
  int ln   = lane & 15;
  int rowBase = (blockIdx.x * 8 + wave) * 16;
  int row = rowBase + ln;
  if (row >= N_CLAUSES) row = N_CLAUSES - 1;   // clamp: EXEC stays all-ones for WMMA

  // A: 16x64 fp32 row -> two 16x32 bf16 fragments, contiguous k order (same as B):
  // half h lane holds K = {16h..16h+15} (then +32 for fragment 1)
  const float4* fr4 = (const float4*)(features + (long)row * N_FEAT);
  int h2 = hv * 4;
  v16us a0u, a1u;
  {
    float4 p0 = fr4[h2 + 0],  p1 = fr4[h2 + 1];
    float4 p2 = fr4[h2 + 2],  p3 = fr4[h2 + 3];
    float4 q0 = fr4[h2 + 8],  q1 = fr4[h2 + 9];
    float4 q2 = fr4[h2 + 10], q3 = fr4[h2 + 11];
#define PK4(dv, b, f) dv[b+0]=f2bf(f.x); dv[b+1]=f2bf(f.y); dv[b+2]=f2bf(f.z); dv[b+3]=f2bf(f.w)
    PK4(a0u, 0, p0); PK4(a0u, 4, p1); PK4(a0u, 8, p2); PK4(a0u, 12, p3);
    PK4(a1u, 0, q0); PK4(a1u, 4, q1); PK4(a1u, 8, q2); PK4(a1u, 12, q3);
#undef PK4
  }
  v16bf A0 = __builtin_bit_cast(v16bf, a0u);
  v16bf A1 = __builtin_bit_cast(v16bf, a1u);

  float part[8];
#pragma unroll
  for (int r = 0; r < 8; ++r) part[r] = 0.f;

#pragma unroll
  for (int t = 0; t < 8; ++t) {
    int N = t * 16 + ln;       // hidden column owned by this lane (C/D layout)
    // B 32x16 bf16: lane holds column N, K = 16h..16h+15 (contiguous in W1t row)
    const unsigned short* bp = &sW1t[N * N_FEAT + hv * 16];
    v16bf B0 = __builtin_bit_cast(v16bf, *(const v16us*)bp);          // K 0..31
    v16bf B1 = __builtin_bit_cast(v16bf, *(const v16us*)(bp + 32));   // K 32..63
    v8f acc = {0.f, 0.f, 0.f, 0.f, 0.f, 0.f, 0.f, 0.f};
    acc = __builtin_amdgcn_wmma_f32_16x16x32_bf16(false, A0, false, B0,
                                                  (short)0, acc, false, false);
    acc = __builtin_amdgcn_wmma_f32_16x16x32_bf16(false, A1, false, B1,
                                                  (short)0, acc, false, false);
    // fused epilogue: +b1, relu, * v[col]; C/D layout: VGPR r -> row r + 8*half
    float2 bvv = sBV[N];
#pragma unroll
    for (int r = 0; r < 8; ++r) {
      float x = acc[r] + bvv.x;
      x = x > 0.f ? x : 0.f;
      part[r] += x * bvv.y;
    }
  }

  // reduce across the 16 lanes (columns) of each half-wave
#pragma unroll
  for (int r = 0; r < 8; ++r) {
    float p = part[r];
    p += __shfl_xor(p, 1, 32);
    p += __shfl_xor(p, 2, 32);
    p += __shfl_xor(p, 4, 32);
    p += __shfl_xor(p, 8, 32);
    part[r] = p;
  }
  float c = ((const float*)hdr)[1];
  if (ln == 0) {
    unsigned localmax = 0u;
#pragma unroll
    for (int r = 0; r < 8; ++r) {
      int ro = rowBase + hv * 8 + r;
      if (ro < N_CLAUSES) {
        float lg = part[r] + c;
        logits[ro] = lg;
        unsigned k = f2key(lg);
        if (k > localmax) localmax = k;
      }
    }
    atomicMax(maxbits, localmax);   // exact -> deterministic
  }
}

// ---------------------------------------------------------------- expl table
__global__ void expl_kernel(const float* __restrict__ logits,
                            const unsigned* hdr,
                            float* __restrict__ expl) {
  int n = blockIdx.x * blockDim.x + threadIdx.x;
  if (n >= N_CLAUSES) return;
  float M = key2f(hdr[0]);
  expl[n] = expf(logits[n] - M);
}

// ---------------------------------------------------------------- proof compaction
// Single block, deterministic stream compaction (fixed order => bitwise-stable sums)
__global__ __launch_bounds__(256)
void compact_kernel(const float* __restrict__ logits,
                    const unsigned char* __restrict__ proof,
                    unsigned* hdr, int* __restrict__ pidx,
                    float* __restrict__ plog) {
  __shared__ unsigned scan[256];
  __shared__ unsigned base;
  int tid = threadIdx.x;
  if (tid == 0) base = 0;
  __syncthreads();
  for (int start = 0; start < N_CLAUSES; start += 256) {
    int n = start + tid;
    unsigned f = (n < N_CLAUSES && proof[n]) ? 1u : 0u;
    scan[tid] = f;
    __syncthreads();
    for (int o = 1; o < 256; o <<= 1) {          // Hillis-Steele inclusive scan
      unsigned v = (tid >= o) ? scan[tid - o] : 0u;
      __syncthreads();
      scan[tid] += v;
      __syncthreads();
    }
    if (f) {
      unsigned pos = base + scan[tid] - 1u;
      pidx[pos] = n;
      plog[pos] = logits[n];
    }
    __syncthreads();
    if (tid == 255) base += scan[255];
    __syncthreads();
  }
  if (tid == 255) hdr[2] = base;
}

// ---------------------------------------------------------------- per-step reduction
// grid = N_STEPS * SPLIT blocks; hot loop reads only sel byte + expl float (5 B/clause)
__global__ __launch_bounds__(256)
void step_kernel(const unsigned char* __restrict__ sel,
                 const float* __restrict__ expl,
                 const int* __restrict__ pidx,
                 const float* __restrict__ plog,
                 const unsigned* hdr,
                 float4* __restrict__ partials) {
  int s   = blockIdx.x / SPLIT;
  int seg = blockIdx.x % SPLIT;
  int tid = threadIdx.x;
  const unsigned char* row = sel + (size_t)s * N_CLAUSES;
  const unsigned* row4 = (const unsigned*)row;
  const float4* ex4 = (const float4*)expl;

  const int chunk = N_CLAUSES / 4 / SPLIT;       // 12500 uints per segment
  const int lo = seg * chunk, hi = lo + chunk;
  float se = 0.f;
  for (int i = lo + tid; i < hi; i += 256) {
    unsigned m = row4[i];
    __builtin_prefetch(&row4[i + 1024], 0, 1);   // speculative -> safe OOB
    float4 ex = ex4[i];
    se += ((m & 0x000000FFu) ? ex.x : 0.f) + ((m & 0x0000FF00u) ? ex.y : 0.f) +
          ((m & 0x00FF0000u) ? ex.z : 0.f) + ((m & 0xFF000000u) ? ex.w : 0.f);
  }

  // proof clauses (~2%): handled entirely by segment 0 via compacted gather list
  float nu = 0.f, ct = 0.f;
  if (seg == 0) {
    int P = (int)hdr[2];
    for (int j = tid; j < P; j += 256) {
      if (row[pidx[j]]) { nu += plog[j]; ct += 1.f; }
    }
  }

  __shared__ float sA[256], sB[256], sC[256];
  sA[tid] = se; sB[tid] = nu; sC[tid] = ct;
  __syncthreads();
  for (int o = 128; o > 0; o >>= 1) {
    if (tid < o) { sA[tid] += sA[tid + o]; sB[tid] += sB[tid + o]; sC[tid] += sC[tid + o]; }
    __syncthreads();
  }
  if (tid == 0) partials[blockIdx.x] = make_float4(sA[0], sB[0], sC[0], 0.f);
}

// ---------------------------------------------------------------- combine + mean
__global__ __launch_bounds__(512)
void final_kernel(const float4* __restrict__ partials, const unsigned* hdr,
                  float* out) {
  __shared__ float sm[512];
  int s = threadIdx.x;                       // one step per thread
  float M = key2f(hdr[0]);
  float se = 0.f, nu = 0.f, ct = 0.f;
#pragma unroll
  for (int seg = 0; seg < SPLIT; ++seg) {    // fixed order -> deterministic
    float4 p = partials[s * SPLIT + seg];
    se += p.x; nu += p.y; ct += p.z;
  }
  sm[s] = (M + logf(se)) - nu / ct;
  __syncthreads();
  for (int o = 256; o > 0; o >>= 1) {
    if (s < o) sm[s] += sm[s + o];
    __syncthreads();
  }
  if (s == 0) out[0] = sm[0] / (float)N_STEPS;
}

// ---------------------------------------------------------------- launch
extern "C" void kernel_launch(void* const* d_in, const int* in_sizes, int n_in,
                              void* d_out, int out_size, void* d_ws, size_t ws_size,
                              hipStream_t stream) {
  const float* features = (const float*)d_in[0];
  const float* W1    = (const float*)d_in[1];
  const float* b1    = (const float*)d_in[2];
  const float* W2    = (const float*)d_in[3];
  const float* b2    = (const float*)d_in[4];
  const float* key_w = (const float*)d_in[5];
  const unsigned char* sel   = (const unsigned char*)d_in[6];   // jax bool = 1 byte
  const unsigned char* proof = (const unsigned char*)d_in[7];

  char* ws = (char*)d_ws;                                 // all offsets 16B aligned
  unsigned*       hdr      = (unsigned*)(ws + 0);         // [0]=max key [1]=c [2]=P
  unsigned short* W1t      = (unsigned short*)(ws + 16);  // 16384 B
  float2*         bv       = (float2*)(ws + 16400);       // 1024 B
  float*          logits   = (float*)(ws + 17424);        // 800000 B
  float*          expl     = (float*)(ws + 817424);       // 800000 B
  int*            pidx     = (int*)(ws + 1617424);        // 800000 B (worst case)
  float*          plog     = (float*)(ws + 2417424);      // 800000 B
  float4*         partials = (float4*)(ws + 3217424);     // 32768 B

  prep_kernel<<<1, 128, 0, stream>>>(W1, b1, W2, b2, key_w, hdr, W1t, bv);
  logits_kernel<<<(N_CLAUSES + 127) / 128, 256, 0, stream>>>(features, W1t, bv,
                                                             hdr, logits, hdr);
  expl_kernel<<<(N_CLAUSES + 255) / 256, 256, 0, stream>>>(logits, hdr, expl);
  compact_kernel<<<1, 256, 0, stream>>>(logits, proof, hdr, pidx, plog);
  step_kernel<<<N_STEPS * SPLIT, 256, 0, stream>>>(sel, expl, pidx, plog, hdr, partials);
  final_kernel<<<1, 512, 0, stream>>>(partials, hdr, (float*)d_out);
}
